// learnable_adaptive_iGCN_4269197492790
// MI455X (gfx1250) — matrix operand order repaired and checked by reading, not compiled, not run on hardware
//
#include <hip/hip_runtime.h>
#include <hip/hip_bf16.h>
#include <math.h>

typedef __attribute__((ext_vector_type(16))) _Float16 v16h;
typedef __attribute__((ext_vector_type(8)))  _Float16 v8h;
typedef __attribute__((ext_vector_type(2)))  _Float16 v2h;
typedef __attribute__((ext_vector_type(8)))  float    v8f;

#define N_NODES 50000   // divisible by 16 -> no row guards in GEMMs
#define N_EDGES 600000
#define IN_DIM  500
#define KPAD    512     // IN_DIM padded to multiple of 32 for WMMA K
#define HID     128
#define OUT_DIM 40
#define OUTPAD  64      // decoder cols padded to 4 column-tiles
#define N_ITERS 8
#define GEMM_BLOCKS (N_NODES / 16)

// ---------------------------------------------------------------------------
// Degree / normalization
// ---------------------------------------------------------------------------
__global__ __launch_bounds__(256) void k_deg_init(float* deg) {
    int i = blockIdx.x * 256 + threadIdx.x;
    if (i < N_NODES) deg[i] = 1.0f;                 // self-loop contribution
}

__global__ __launch_bounds__(256) void k_deg_count(const int* __restrict__ dst,
                                                   float* __restrict__ deg) {
    int e = blockIdx.x * 256 + threadIdx.x;
    if (e < N_EDGES) atomicAdd(&deg[dst[e]], 1.0f);
}

__global__ __launch_bounds__(256) void k_dinv(float* deg) {
    int i = blockIdx.x * 256 + threadIdx.x;
    if (i < N_NODES) deg[i] = rsqrtf(fmaxf(deg[i], 1e-12f));
}

// ---------------------------------------------------------------------------
// Weight conversion f32 -> f16, TRANSPOSED to [col][K] so B fragments are
// two separate 16-byte runs per lane (two global_load_b128).  Zero padding baked in.
// ---------------------------------------------------------------------------
__global__ __launch_bounds__(256) void k_convert_weights(
        const float* __restrict__ Wenc, const float* __restrict__ Wgc,
        const float* __restrict__ Wdec,
        _Float16* __restrict__ WencT,   // [HID][KPAD]
        _Float16* __restrict__ WgcT,    // [HID][HID]
        _Float16* __restrict__ WdecT) { // [OUTPAD][HID]
    int t = blockIdx.x * 256 + threadIdx.x;
    if (t < HID * KPAD) {
        int c = t >> 9, k = t & (KPAD - 1);
        WencT[t] = (_Float16)((k < IN_DIM) ? Wenc[k * HID + c] : 0.0f);
    }
    if (t < HID * HID) {
        int c = t >> 7, k = t & (HID - 1);
        WgcT[t] = (_Float16)Wgc[k * HID + c];
    }
    if (t < OUTPAD * HID) {
        int c = t >> 7, k = t & (HID - 1);
        WdecT[t] = (_Float16)((c < OUT_DIM) ? Wdec[k * OUT_DIM + c] : 0.0f);
    }
}

// ---------------------------------------------------------------------------
// Encoder: h = relu(x @ Wenc + b_enc); also emit f16 copy of h.
// Block = 256 threads (8 waves), tile = 16 rows x 128 cols, wave w -> col tile.
// Double-buffered LDS A-tile: one barrier per K-step, next x float2 prefetched
// before the barrier so the global load overlaps barrier + WMMA.
// ---------------------------------------------------------------------------
__global__ __launch_bounds__(256) void k_encoder(
        const float* __restrict__ x, const _Float16* __restrict__ WT,
        const float* __restrict__ bias,
        float* __restrict__ h, _Float16* __restrict__ h16) {
    __shared__ _Float16 sA[2][16 * 32];              // double-buffered [row][k]
    const int row0 = blockIdx.x << 4;
    const int tid  = threadIdx.x;
    const int wave = tid >> 5;
    const int lane = tid & 31;
    const int mrow   = lane & 15;
    const int kbaseA = (lane >> 4) << 3;             // 0 or 8
    const int kbaseB = (lane >> 4) << 3;             // 0 or 8
    const int col    = (wave << 4) + (lane & 15);
    // A-loader mapping: thread t loads one float2 of row r at halves [seg,seg+1]
    const int r   = tid >> 4;
    const int seg = (tid & 15) << 1;
    const float*    xrow = x  + (size_t)(row0 + r) * IN_DIM + seg;
    const _Float16* wcol = WT + (size_t)col * KPAD + kbaseB;

    v8f acc = {};
    float2 v = *(const float2*)(xrow);               // K-step 0 always in range
    int ph = 0;
#pragma unroll 4
    for (int k = 0; k < KPAD / 32; ++k) {
        *(v2h*)&sA[ph][(r << 5) + seg] = (v2h){ (_Float16)v.x, (_Float16)v.y };
        if (k < KPAD / 32 - 1) {                     // prefetch next step
            int k1 = (k + 1) << 5;
            v = make_float2(0.0f, 0.0f);
            if (k1 < 480 || seg < 20)                // K-tail guard (500..511 -> 0)
                v = *(const float2*)(xrow + k1);
        }
        __syncthreads();
        const _Float16* sa = &sA[ph][(mrow << 5) + kbaseA];
        const v8h a0 = *(const v8h*)(sa);
        const v8h a1 = *(const v8h*)(sa + 16);
        const v8h b0 = *(const v8h*)(wcol + (k << 5));
        const v8h b1 = *(const v8h*)(wcol + (k << 5) + 16);
        const v16h a = __builtin_shufflevector(a0, a1, 0,1,2,3,4,5,6,7,8,9,10,11,12,13,14,15);
        const v16h b = __builtin_shufflevector(b0, b1, 0,1,2,3,4,5,6,7,8,9,10,11,12,13,14,15);
        acc = __builtin_amdgcn_wmma_f32_16x16x32_f16(
                  false, a, false, b, (short)0, acc, false, false);
        ph ^= 1;
    }
    const int mbase = (lane >> 4) << 3;
    const float bc = bias[col];
#pragma unroll
    for (int q = 0; q < 8; ++q) {
        int grow = row0 + mbase + q;
        float vv = fmaxf(acc[q] + bc, 0.0f);
        h  [(size_t)grow * HID + col] = vv;
        h16[(size_t)grow * HID + col] = (_Float16)vv;
    }
}

// ---------------------------------------------------------------------------
// GC linear: m = h16 @ WgcT^T.  A and B both f16 in global -> no LDS, no
// barriers; 4 fully unrolled WMMA steps, all fragments via global_load_b128.
// ---------------------------------------------------------------------------
__global__ __launch_bounds__(256) void k_gemm_gc(
        const _Float16* __restrict__ A, const _Float16* __restrict__ WT,
        float* __restrict__ m) {
    const int row0 = blockIdx.x << 4;
    const int wave = threadIdx.x >> 5;
    const int lane = threadIdx.x & 31;
    const int mrow   = lane & 15;
    const int kbaseA = (lane >> 4) << 3;
    const int kbaseB = (lane >> 4) << 3;
    const int col    = (wave << 4) + (lane & 15);
    const _Float16* arow = A  + (size_t)(row0 + mrow) * HID + kbaseA;
    const _Float16* wcol = WT + (size_t)col * HID + kbaseB;

    v8f acc = {};
#pragma unroll
    for (int k0 = 0; k0 < HID; k0 += 32) {
        const v8h a0 = *(const v8h*)(arow + k0);
        const v8h a1 = *(const v8h*)(arow + k0 + 16);
        const v8h b0 = *(const v8h*)(wcol + k0);
        const v8h b1 = *(const v8h*)(wcol + k0 + 16);
        const v16h a = __builtin_shufflevector(a0, a1, 0,1,2,3,4,5,6,7,8,9,10,11,12,13,14,15);
        const v16h b = __builtin_shufflevector(b0, b1, 0,1,2,3,4,5,6,7,8,9,10,11,12,13,14,15);
        acc = __builtin_amdgcn_wmma_f32_16x16x32_f16(
                  false, a, false, b, (short)0, acc, false, false);
    }
    const int mbase = (lane >> 4) << 3;
#pragma unroll
    for (int q = 0; q < 8; ++q)
        m[(size_t)(row0 + mbase + q) * HID + col] = acc[q];
}

// ---------------------------------------------------------------------------
// agg = b_gc + dinv[i]^2 * m[i]   (self-loop term, single writer per element)
// node id forced into an SGPR -> dinv load scalarizes (one fetch per wave).
// ---------------------------------------------------------------------------
__global__ __launch_bounds__(256) void k_agg_init(
        const float* __restrict__ dinv, const float* __restrict__ m,
        const float* __restrict__ bgc, float* __restrict__ agg) {
    int i = __builtin_amdgcn_readfirstlane(blockIdx.x * 8 + (threadIdx.x >> 5));
    if (i >= N_NODES) return;
    int c0 = (threadIdx.x & 31) << 2;
    float di = dinv[i];
    float w = di * di;
    float4 mv = *(const float4*)(m + (size_t)i * HID + c0);
    float4 b4 = *(const float4*)(bgc + c0);
    float4 r;
    r.x = b4.x + w * mv.x; r.y = b4.y + w * mv.y;
    r.z = b4.z + w * mv.z; r.w = b4.w + w * mv.w;
    *(float4*)(agg + (size_t)i * HID + c0) = r;
}

// ---------------------------------------------------------------------------
// Edge scatter: agg[dst] += dinv[src]*dinv[dst] * m[src]
// One wave per edge, float4 per lane = full 512B row.  Edge id forced uniform
// so src/dst/dinv become scalar loads.  m+agg = 51MB -> L2-resident atomics.
// ---------------------------------------------------------------------------
__global__ __launch_bounds__(256) void k_edge_scatter(
        const int* __restrict__ src, const int* __restrict__ dst,
        const float* __restrict__ dinv, const float* __restrict__ m,
        float* __restrict__ agg) {
    int e = __builtin_amdgcn_readfirstlane(blockIdx.x * 8 + (threadIdx.x >> 5));
    if (e >= N_EDGES) return;
    int lane = threadIdx.x & 31;
    int s = src[e], d = dst[e];
    float w = dinv[s] * dinv[d];
    float4 mv = *(const float4*)(m + (size_t)s * HID + (lane << 2));
    float* ap = agg + (size_t)d * HID + (lane << 2);
    atomicAdd(ap + 0, w * mv.x);
    atomicAdd(ap + 1, w * mv.y);
    atomicAdd(ap + 2, w * mv.z);
    atomicAdd(ap + 3, w * mv.w);
}

// ---------------------------------------------------------------------------
// h = sf*h + (1-sf)*relu(agg);  sf = sigmoid(schedule[it]); refresh h16.
// ---------------------------------------------------------------------------
__global__ __launch_bounds__(256) void k_blend(
        const float* __restrict__ sched, int it,
        const float* __restrict__ agg, float* __restrict__ h,
        _Float16* __restrict__ h16) {
    int i = __builtin_amdgcn_readfirstlane(blockIdx.x * 8 + (threadIdx.x >> 5));
    if (i >= N_NODES) return;
    int c0 = (threadIdx.x & 31) << 2;
    float s  = sched[it];
    float sf = 1.0f / (1.0f + expf(-s));
    float4 a4 = *(const float4*)(agg + (size_t)i * HID + c0);
    float4 h4 = *(const float4*)(h   + (size_t)i * HID + c0);
    float4 r;
    r.x = sf * h4.x + (1.0f - sf) * fmaxf(a4.x, 0.0f);
    r.y = sf * h4.y + (1.0f - sf) * fmaxf(a4.y, 0.0f);
    r.z = sf * h4.z + (1.0f - sf) * fmaxf(a4.z, 0.0f);
    r.w = sf * h4.w + (1.0f - sf) * fmaxf(a4.w, 0.0f);
    *(float4*)(h + (size_t)i * HID + c0) = r;
    *(v2h*)(h16 + (size_t)i * HID + c0)     = (v2h){ (_Float16)r.x, (_Float16)r.y };
    *(v2h*)(h16 + (size_t)i * HID + c0 + 2) = (v2h){ (_Float16)r.z, (_Float16)r.w };
}

// ---------------------------------------------------------------------------
// Decoder: out = h16 @ WdecT^T + b_dec.  No LDS, no barriers.
// Block = 128 threads (4 waves) covering 64 padded cols; cols>=40 masked.
// ---------------------------------------------------------------------------
__global__ __launch_bounds__(128) void k_decoder(
        const _Float16* __restrict__ A, const _Float16* __restrict__ WT,
        const float* __restrict__ bias, float* __restrict__ out) {
    const int row0 = blockIdx.x << 4;
    const int wave = threadIdx.x >> 5;
    const int lane = threadIdx.x & 31;
    const int mrow   = lane & 15;
    const int kbaseA = (lane >> 4) << 3;
    const int kbaseB = (lane >> 4) << 3;
    const int col    = (wave << 4) + (lane & 15);
    const _Float16* arow = A  + (size_t)(row0 + mrow) * HID + kbaseA;
    const _Float16* wcol = WT + (size_t)col * HID + kbaseB;

    v8f acc = {};
#pragma unroll
    for (int k0 = 0; k0 < HID; k0 += 32) {
        const v8h a0 = *(const v8h*)(arow + k0);
        const v8h a1 = *(const v8h*)(arow + k0 + 16);
        const v8h b0 = *(const v8h*)(wcol + k0);
        const v8h b1 = *(const v8h*)(wcol + k0 + 16);
        const v16h a = __builtin_shufflevector(a0, a1, 0,1,2,3,4,5,6,7,8,9,10,11,12,13,14,15);
        const v16h b = __builtin_shufflevector(b0, b1, 0,1,2,3,4,5,6,7,8,9,10,11,12,13,14,15);
        acc = __builtin_amdgcn_wmma_f32_16x16x32_f16(
                  false, a, false, b, (short)0, acc, false, false);
    }
    const int mbase = (lane >> 4) << 3;
    const float bc = (col < OUT_DIM) ? bias[col] : 0.0f;
#pragma unroll
    for (int q = 0; q < 8; ++q) {
        int grow = row0 + mbase + q;
        if (col < OUT_DIM)
            out[(size_t)grow * OUT_DIM + col] = acc[q] + bc;
    }
}

// ---------------------------------------------------------------------------
extern "C" void kernel_launch(void* const* d_in, const int* in_sizes, int n_in,
                              void* d_out, int out_size, void* d_ws, size_t ws_size,
                              hipStream_t stream) {
    (void)in_sizes; (void)n_in; (void)out_size; (void)ws_size;
    const float* x    = (const float*)d_in[0];
    const int*   ei   = (const int*)  d_in[1];   // [2][E] (src row, dst row)
    const float* Wenc = (const float*)d_in[2];
    const float* benc = (const float*)d_in[3];
    const float* Wgc  = (const float*)d_in[4];
    const float* bgc  = (const float*)d_in[5];
    const float* Wdec = (const float*)d_in[6];
    const float* bdec = (const float*)d_in[7];
    const float* schd = (const float*)d_in[8];
    float* out = (float*)d_out;

    char* ws = (char*)d_ws;
    size_t off = 0;
    auto alloc = [&](size_t bytes) -> void* {
        void* p = (void*)(ws + off);
        off = (off + bytes + 255) & ~(size_t)255;
        return p;
    };
    float*    dinv  = (float*)   alloc(sizeof(float)    * N_NODES);
    float*    h     = (float*)   alloc(sizeof(float)    * (size_t)N_NODES * HID);
    _Float16* h16   = (_Float16*)alloc(sizeof(_Float16) * (size_t)N_NODES * HID);
    float*    m     = (float*)   alloc(sizeof(float)    * (size_t)N_NODES * HID);
    float*    agg   = (float*)   alloc(sizeof(float)    * (size_t)N_NODES * HID);
    _Float16* WencT = (_Float16*)alloc(sizeof(_Float16) * HID * KPAD);
    _Float16* WgcT  = (_Float16*)alloc(sizeof(_Float16) * HID * HID);
    _Float16* WdecT = (_Float16*)alloc(sizeof(_Float16) * OUTPAD * HID);

    const int* src = ei;
    const int* dst = ei + N_EDGES;

    k_deg_init <<<(N_NODES + 255) / 256, 256, 0, stream>>>(dinv);
    k_deg_count<<<(N_EDGES + 255) / 256, 256, 0, stream>>>(dst, dinv);
    k_dinv     <<<(N_NODES + 255) / 256, 256, 0, stream>>>(dinv);
    k_convert_weights<<<(HID * KPAD + 255) / 256, 256, 0, stream>>>(
        Wenc, Wgc, Wdec, WencT, WgcT, WdecT);
    k_encoder<<<GEMM_BLOCKS, 256, 0, stream>>>(x, WencT, benc, h, h16);

    for (int it = 0; it < N_ITERS; ++it) {
        k_gemm_gc     <<<GEMM_BLOCKS, 256, 0, stream>>>(h16, WgcT, m);
        k_agg_init    <<<N_NODES / 8, 256, 0, stream>>>(dinv, m, bgc, agg);
        k_edge_scatter<<<N_EDGES / 8, 256, 0, stream>>>(src, dst, dinv, m, agg);
        k_blend       <<<N_NODES / 8, 256, 0, stream>>>(schd, it, agg, h, h16);
    }

    k_decoder<<<GEMM_BLOCKS, 128, 0, stream>>>(h16, WdecT, bdec, out);
}
